// MRICls_73023033966684
// MI455X (gfx1250) — compile-verified, matrix operand order checked by reading, not run on hardware
//
#include <hip/hip_runtime.h>
#include <hip/hip_bf16.h>

// Types matching the gfx1250 WMMA builtin signatures
typedef __attribute__((ext_vector_type(16))) __bf16 v16bf;
typedef __attribute__((ext_vector_type(8)))  float  v8f;
typedef __attribute__((ext_vector_type(4)))  float  fvec4;

union FragU { v16bf bf; fvec4 f4[2]; };

// Load one lane's 16 bf16 A/B-fragment values: K sub-block {0..7} at p,
// K sub-block {16..23} at p+16 elements (per ISA 7.12.2 16-bit layout).
// Caller pre-offsets p by +8 elements for lanes 16..31.
__device__ __forceinline__ v16bf frag_ld(const __bf16* p) {
    FragU u;
    u.f4[0] = *(const fvec4*)(p);
    u.f4[1] = *(const fvec4*)(p + 16);
    return u.bf;
}

#define WMMA_BF16(a, b, c) \
    __builtin_amdgcn_wmma_f32_16x16x32_bf16(false, (a), false, (b), (short)0, (c), false, false)

// J=116, feat dim = 464 (padded to 480 = 15 k-steps of 32)
// Block: 128 threads = 4 waves; 64 samples per block (16 per wave).
__global__ __launch_bounds__(128) void mricls_kernel(
    const float* __restrict__ F,  const float* __restrict__ S,
    const float* __restrict__ W1, const float* __restrict__ b1,
    const float* __restrict__ W2, const float* __restrict__ b2,
    const float* __restrict__ W3, const float* __restrict__ b3,
    float* __restrict__ out)
{
    __shared__ __bf16 feat[64 * 480];   // [sample][k] gated features, bf16
    __shared__ __bf16 w1t [64 * 480];   // [n][k] W1 transposed, zero-padded K
    __shared__ __bf16 w2t [32 * 64];    // [n][k] W2 transposed
    __shared__ __bf16 h1  [64 * 64];    // layer-1 activations
    __shared__ __bf16 h2  [64 * 32];    // layer-2 activations
    __shared__ float  b1s[64];
    __shared__ float  b2s[32];
    __shared__ float  w3s[64];          // W3 (32x2) row-major
    __shared__ float  b3s[2];

    const int tid = threadIdx.x;
    const int s0  = blockIdx.x * 64;    // first sample of this block

    // ---- Cooperative LDS fill -------------------------------------------
    if (tid < 64) { b1s[tid] = b1[tid]; w3s[tid] = W3[tid]; }
    if (tid < 32) { b2s[tid] = b2[tid]; }
    if (tid < 2)  { b3s[tid] = b3[tid]; }

    // W1 (464x64, row-major) -> w1t[n][k] bf16; pad K 464..479 with zeros.
    for (int idx = tid; idx < 64 * 464; idx += 128) {
        int n = idx / 464;
        int k = idx - n * 464;
        w1t[n * 480 + k] = (__bf16)W1[k * 64 + n];
    }
    for (int idx = tid; idx < 64 * 16; idx += 128) {
        int n = idx >> 4;
        w1t[n * 480 + 464 + (idx & 15)] = (__bf16)0.0f;
    }
    // W2 (64x32) -> w2t[n][k]
    for (int idx = tid; idx < 32 * 64; idx += 128) {
        int n = idx >> 6, k = idx & 63;
        w2t[n * 64 + k] = (__bf16)W2[k * 32 + n];
    }

    // Gating + feature build: coalesced streaming of F and S.
    for (int s = 0; s < 64; ++s) {
        if (tid < 116) {
            const size_t base = (size_t)(s0 + s) * 232;
            float f0 = F[base + tid];
            float f1 = F[base + 116 + tid];
            float e0 = S[base + tid];
            float e1 = S[base + 116 + tid];
            // g = sigmoid(mean(F,ch) * mean(S,ch)) = sigmoid(0.25*(f0+f1)*(e0+e1))
            float g = 1.0f / (1.0f + __expf(-0.25f * (f0 + f1) * (e0 + e1)));
            __bf16* fr = &feat[s * 480];
            fr[tid]       = (__bf16)(f0 * g);
            fr[116 + tid] = (__bf16)(f1 * g);
            fr[232 + tid] = (__bf16)(e0 * g);
            fr[348 + tid] = (__bf16)(e1 * g);
        }
        if (tid < 16) feat[s * 480 + 464 + tid] = (__bf16)0.0f;
    }
    __syncthreads();

    // ---- Per-wave WMMA: each wave owns 16 samples -----------------------
    const int lane = tid & 31;
    const int col  = lane & 15;                 // N/column within 16
    const int ws0  = (tid >> 5) * 16;           // wave's first sample
    const int sub  = (lane & 16) ? 8 : 0;       // K sub-block for hi lanes
    const int mhi  = (lane & 16) ? 8 : 0;       // C/D row offset for hi lanes

    // Layer 1: (16 x 480) x (480 x 64), 15 k-steps x 4 N-tiles
    v8f a0 = {}, a1 = {}, a2 = {}, a3 = {};
    const __bf16* arow = &feat[(ws0 + col) * 480 + sub];
    const __bf16* br0  = &w1t [( 0 + col) * 480 + sub];
    const __bf16* br1  = &w1t [(16 + col) * 480 + sub];
    const __bf16* br2  = &w1t [(32 + col) * 480 + sub];
    const __bf16* br3  = &w1t [(48 + col) * 480 + sub];
    for (int k = 0; k < 480; k += 32) {
        v16bf a = frag_ld(arow + k);
        a0 = WMMA_BF16(a, frag_ld(br0 + k), a0);
        a1 = WMMA_BF16(a, frag_ld(br1 + k), a1);
        a2 = WMMA_BF16(a, frag_ld(br2 + k), a2);
        a3 = WMMA_BF16(a, frag_ld(br3 + k), a3);
    }
    // bias + ReLU, accumulator (VGPR r -> row, lane -> col) -> h1 bf16
    {
        float bv0 = b1s[ 0 + col];
        float bv1 = b1s[16 + col];
        float bv2 = b1s[32 + col];
        float bv3 = b1s[48 + col];
        #pragma unroll
        for (int r = 0; r < 8; ++r) {
            int row = (ws0 + mhi + r) * 64;
            h1[row +  0 + col] = (__bf16)fmaxf(a0[r] + bv0, 0.0f);
            h1[row + 16 + col] = (__bf16)fmaxf(a1[r] + bv1, 0.0f);
            h1[row + 32 + col] = (__bf16)fmaxf(a2[r] + bv2, 0.0f);
            h1[row + 48 + col] = (__bf16)fmaxf(a3[r] + bv3, 0.0f);
        }
    }
    __syncthreads();

    // Layer 2: (16 x 64) x (64 x 32), 2 k-steps x 2 N-tiles
    v8f c0 = {}, c1 = {};
    const __bf16* arow2 = &h1[(ws0 + col) * 64 + sub];
    const __bf16* wr0   = &w2t[( 0 + col) * 64 + sub];
    const __bf16* wr1   = &w2t[(16 + col) * 64 + sub];
    #pragma unroll
    for (int k = 0; k < 64; k += 32) {
        v16bf a = frag_ld(arow2 + k);
        c0 = WMMA_BF16(a, frag_ld(wr0 + k), c0);
        c1 = WMMA_BF16(a, frag_ld(wr1 + k), c1);
    }
    {
        float bv0 = b2s[ 0 + col];
        float bv1 = b2s[16 + col];
        #pragma unroll
        for (int r = 0; r < 8; ++r) {
            int row = (ws0 + mhi + r) * 32;
            h2[row +  0 + col] = (__bf16)fmaxf(c0[r] + bv0, 0.0f);
            h2[row + 16 + col] = (__bf16)fmaxf(c1[r] + bv1, 0.0f);
        }
    }
    __syncthreads();

    // Layer 3 (32 -> 2) + 2-class softmax, one thread per sample
    if (tid < 64) {
        const __bf16* hp = &h2[tid * 32];
        float l0 = b3s[0], l1 = b3s[1];
        #pragma unroll
        for (int n = 0; n < 32; ++n) {
            float hv = (float)hp[n];
            l0 += hv * w3s[2 * n];
            l1 += hv * w3s[2 * n + 1];
        }
        float p0 = 1.0f / (1.0f + __expf(l1 - l0));   // softmax over 2 classes
        size_t o = (size_t)(s0 + tid) * 2;
        out[o]     = p0;
        out[o + 1] = 1.0f - p0;
    }
}

extern "C" void kernel_launch(void* const* d_in, const int* in_sizes, int n_in,
                              void* d_out, int out_size, void* d_ws, size_t ws_size,
                              hipStream_t stream) {
    const float* F  = (const float*)d_in[0];
    const float* S  = (const float*)d_in[1];
    const float* W1 = (const float*)d_in[2];
    const float* b1 = (const float*)d_in[3];
    const float* W2 = (const float*)d_in[4];
    const float* b2 = (const float*)d_in[5];
    const float* W3 = (const float*)d_in[6];
    const float* b3 = (const float*)d_in[7];
    float* out = (float*)d_out;

    const int Btot = in_sizes[0] / 232;   // B*2*116 / (2*116)
    const int grid = Btot / 64;           // 64 samples per block

    mricls_kernel<<<grid, 128, 0, stream>>>(F, S, W1, b1, W2, b2, W3, b3, out);
}
